// MultiHeadAttention_2671469658700
// MI455X (gfx1250) — matrix operand order repaired and checked by reading, not compiled, not run on hardware
//
#include <hip/hip_runtime.h>
#include <hip/hip_bf16.h>

typedef __attribute__((ext_vector_type(16))) _Float16 v16h;
typedef __attribute__((ext_vector_type(8)))  float    v8f;
typedef unsigned int u32;
typedef __attribute__((ext_vector_type(4))) unsigned int u32x4;
typedef __attribute__((ext_vector_type(8))) int         i32x8;
typedef __attribute__((ext_vector_type(4))) int         i32x4;

union Frag16 { u32 u[8]; v16h h; };
union HPack  { _Float16 h[2]; u32 u; };

#define BSZ    4
#define S_LEN  2048
#define DMODEL 1024
#define NHEAD  16
#define DHEAD  64
#define MROWS  (BSZ * S_LEN)          // 8192
#define MASKV  (-1000000.0f)

// ---------------------------------------------------------------------------
// Tiled WMMA GEMM: C[M,N] = A[M,K] * B[K,N], f16 compute / f32 accumulate.
// Block = 256 threads (8 waves), 128x128 C tile, K-step = 32.
// ---------------------------------------------------------------------------
#define AROW 40   // padded LDS row (32 f16 + 8 pad) -> bank-spread dword loads

__device__ inline void stage_A_tile(const float* __restrict__ A, _Float16* As,
                                    int tid, int K, long base) {
  #pragma unroll
  for (int i = 0; i < 4; ++i) {
    const int vid = tid + i * 256;          // 1024 float4 = 128x32 f32
    const int r = vid >> 3, c = (vid & 7) * 4;
    const float4 a4 = *(const float4*)(A + (long)r * K + base + c);
    HPack p0, p1;
    p0.h[0] = (_Float16)a4.x; p0.h[1] = (_Float16)a4.y;
    p1.h[0] = (_Float16)a4.z; p1.h[1] = (_Float16)a4.w;
    *(u32*)&As[r * AROW + c]     = p0.u;
    *(u32*)&As[r * AROW + c + 2] = p1.u;
  }
}

__device__ inline void stage_A_tile(const _Float16* __restrict__ A, _Float16* As,
                                    int tid, int K, long base) {
  #pragma unroll
  for (int i = 0; i < 4; ++i) {
    const int vid = tid + i * 256;          // 1024 uint2 = 128x32 f16
    const int r = vid >> 3, c = (vid & 7) * 4;
    const uint2 a2 = *(const uint2*)(A + (long)r * K + base + c);
    *(u32*)&As[r * AROW + c]     = a2.x;
    *(u32*)&As[r * AROW + c + 2] = a2.y;
  }
}

template <typename TA, typename TO>
__global__ __launch_bounds__(256) void gemm_wmma_kernel(
    const TA* __restrict__ A, const float* __restrict__ Bw, TO* __restrict__ C,
    int M, int N, int K) {
  __shared__ _Float16 As[128 * AROW];   // A tile, row-major [m][k]
  __shared__ _Float16 Bs[128 * AROW];   // B tile, TRANSPOSED [n][k]

  const int tid  = threadIdx.x;
  const int wave = tid >> 5, lane = tid & 31;
  const int lrow = lane & 15, lhalf = lane >> 4;
  const long bm = (long)blockIdx.y * 128, bn = (long)blockIdx.x * 128;
  const int wm = (wave >> 1) * 32, wn = (wave & 1) * 64;

  v8f acc[2][4];
  #pragma unroll
  for (int m = 0; m < 2; ++m)
    #pragma unroll
    for (int n = 0; n < 4; ++n) acc[m][n] = (v8f){};

  for (int k0 = 0; k0 < K; k0 += 32) {
    stage_A_tile(A, As, tid, K, bm * K + k0);
    // B tile 32x128 f32 -> Bs[n][k] f16 (transposed so B-frag pairs are contiguous)
    #pragma unroll
    for (int i = 0; i < 4; ++i) {
      const int vid = tid + i * 256;
      const int r = vid >> 5, c = (vid & 31) * 4;
      const float4 b4 = *(const float4*)(Bw + (long)(k0 + r) * N + bn + c);
      Bs[(c + 0) * AROW + r] = (_Float16)b4.x;
      Bs[(c + 1) * AROW + r] = (_Float16)b4.y;
      Bs[(c + 2) * AROW + r] = (_Float16)b4.z;
      Bs[(c + 3) * AROW + r] = (_Float16)b4.w;
    }
    __syncthreads();

    Frag16 afr[2];
    #pragma unroll
    for (int m = 0; m < 2; ++m) {
      const int row = wm + m * 16 + lrow;
      #pragma unroll
      for (int v = 0; v < 8; ++v) {   // ISA 16-bit A 16x32 layout
        const int kk = (v < 4 ? 2 * v : 16 + 2 * (v - 4)) + lhalf * 8;
        afr[m].u[v] = *(const u32*)&As[row * AROW + kk];
      }
    }
    #pragma unroll
    for (int nf = 0; nf < 4; ++nf) {
      Frag16 bfr;
      const int col = wn + nf * 16 + lrow;
      #pragma unroll
      for (int v = 0; v < 8; ++v) {   // B 32x16: same k order as A (ISA table layout)
        const int kk = (v < 4 ? 2 * v : 16 + 2 * (v - 4)) + lhalf * 8;
        bfr.u[v] = *(const u32*)&Bs[col * AROW + kk];
      }
      #pragma unroll
      for (int m = 0; m < 2; ++m)
        acc[m][nf] = __builtin_amdgcn_wmma_f32_16x16x32_f16(
            false, afr[m].h, false, bfr.h, (short)0, acc[m][nf], false, false);
    }
    __syncthreads();
  }

  #pragma unroll
  for (int m = 0; m < 2; ++m) {
    const long row0 = bm + wm + m * 16 + lhalf * 8;  // C/D layout: VGPR r -> M = r (+8 hi half)
    #pragma unroll
    for (int nf = 0; nf < 4; ++nf) {
      const long col = bn + wn + nf * 16 + lrow;
      #pragma unroll
      for (int r = 0; r < 8; ++r)
        C[(row0 + r) * N + col] = (TO)acc[m][nf][r];
    }
  }
}

// ---------------------------------------------------------------------------
// Flash attention: block = (b, h, 64-row Q tile), 128 threads = 4 waves,
// each wave owns 16 query rows. KV swept in 64-wide steps; scores never
// touch HBM. K tiles staged by the Tensor Data Mover (TDM): 64x64 f16 2D
// tile with D#-generated LDS padding matching the FROW=72 layout.
// ---------------------------------------------------------------------------
#define FROW 72   // padded LDS row of 64 f16 (= 32 DW data + 4 DW pad)

__global__ __launch_bounds__(128) void flash_attn_kernel(
    const _Float16* __restrict__ Qp, const _Float16* __restrict__ Kp,
    const _Float16* __restrict__ Vp, const int* __restrict__ valid_lens,
    _Float16* __restrict__ Op) {
  __shared__ _Float16 Qs[64 * FROW];        // [q][d]
  __shared__ _Float16 Ks[64 * FROW];        // [kv][d]  (B-frag for Q*K^T)
  __shared__ _Float16 VsT[64 * FROW];       // [d][kv]  (B-frag for P*V)
  __shared__ _Float16 Ps[4][16 * FROW];     // per-wave P relayout scratch

  const int tid  = threadIdx.x;
  const int wave = tid >> 5, lane = tid & 31;
  const int lrow = lane & 15, lhalf = lane >> 4;
  const int b = blockIdx.y / NHEAD, h = blockIdx.y % NHEAD;
  const int q0 = blockIdx.x * 64;
  const int vlen = valid_lens[b];
  const long headoff = (long)h * DHEAD;

  // stage Q tile (64x64 f16)
  #pragma unroll
  for (int i = 0; i < 8; ++i) {
    const int vid = tid + i * 128;
    const int r = vid >> 4, c = (vid & 15) * 4;
    const uint2 q2 = *(const uint2*)(Qp + (long)(b * S_LEN + q0 + r) * DMODEL + headoff + c);
    *(u32*)&Qs[r * FROW + c]     = q2.x;
    *(u32*)&Qs[r * FROW + c + 2] = q2.y;
  }
  __syncthreads();

  Frag16 qa[2];                              // Q A-frags, k0 = 0 and 32
  #pragma unroll
  for (int kf = 0; kf < 2; ++kf)
    #pragma unroll
    for (int v = 0; v < 8; ++v) {
      const int kk = kf * 32 + (v < 4 ? 2 * v : 16 + 2 * (v - 4)) + lhalf * 8;
      qa[kf].u[v] = *(const u32*)&Qs[(wave * 16 + lrow) * FROW + kk];
    }

  v8f oacc[4];
  #pragma unroll
  for (int df = 0; df < 4; ++df) oacc[df] = (v8f){};
  float mst[8], lst[8];
  #pragma unroll
  for (int r = 0; r < 8; ++r) { mst[r] = -INFINITY; lst[r] = 0.0f; }

  for (int kv0 = 0; kv0 < S_LEN; kv0 += 64) {
    __syncthreads();                         // WAR on Ks/VsT

    // ---- K tile via Tensor Data Mover (wave 0 issues the DMA) ----
    if (wave == 0) {
      const unsigned long long ga =
          (unsigned long long)(Kp + (long)(b * S_LEN + kv0) * DMODEL + headoff);
      u32x4 g0;
      g0[0] = 1u;                                        // count=1 valid descriptor
      g0[1] = (u32)(unsigned long long)&Ks[0];           // lds_addr (low 32b of generic)
      g0[2] = (u32)(ga & 0xFFFFFFFFull);                 // global_addr[31:0]
      g0[3] = (u32)((ga >> 32) & 0x1FFFFFFull) | (2u << 30);  // addr[56:32], type=2
      i32x8 g1;
      g1[0] = (int)((1u << 16)      // data_size = 2 bytes
                  | (1u << 20)      // pad_enable
                  | (4u << 22)      // pad_interval: every 32 DWORDs
                  | (3u << 25));    // pad_amount: 4 DWORDs (-> 72-halfword rows)
      g1[1] = (int)(64u << 16);     // tensor_dim0 = 64
      g1[2] = (int)(64u << 16);     // tensor_dim1 = 64
      g1[3] = (int)(64u << 16);     // tile_dim0 = 64
      g1[4] = 64;                   // tile_dim1 = 64, tile_dim2 = 0
      g1[5] = DMODEL;               // tensor_dim0_stride = 1024 elements
      g1[6] = 0;
      g1[7] = 0;
      const i32x4 gz4 = (i32x4){0, 0, 0, 0};             // 2D tensor: groups 2/3 unused
      const i32x8 gz8 = (i32x8){0, 0, 0, 0, 0, 0, 0, 0};
      __builtin_amdgcn_tensor_load_to_lds(g0, g1, gz4, gz4, gz8, 0);
      __builtin_amdgcn_s_wait_tensorcnt(0);
    }

    // ---- V tile staged manually (needs transpose, TDM can't do that) ----
    #pragma unroll
    for (int i = 0; i < 8; ++i) {
      const int vid = tid + i * 128;
      const int r = vid >> 4, c = (vid & 15) * 4;
      const long g = (long)(b * S_LEN + kv0 + r) * DMODEL + headoff + c;
      const uint2 v2 = *(const uint2*)(Vp + g);
      HPack t0, t1; t0.u = v2.x; t1.u = v2.y;
      VsT[(c + 0) * FROW + r] = t0.h[0];
      VsT[(c + 1) * FROW + r] = t0.h[1];
      VsT[(c + 2) * FROW + r] = t1.h[0];
      VsT[(c + 3) * FROW + r] = t1.h[1];
    }
    __syncthreads();

    // S = Q * K^T  (16 x 64 per wave)
    v8f sacc[4];
    #pragma unroll
    for (int nf = 0; nf < 4; ++nf) sacc[nf] = (v8f){};
    #pragma unroll
    for (int nf = 0; nf < 4; ++nf)
      #pragma unroll
      for (int kf = 0; kf < 2; ++kf) {
        Frag16 bfr;
        #pragma unroll
        for (int v = 0; v < 8; ++v) {
          const int kk = kf * 32 + (v < 4 ? 2 * v : 16 + 2 * (v - 4)) + lhalf * 8;   // B[d][j] = K[j][d]
          bfr.u[v] = *(const u32*)&Ks[(nf * 16 + lrow) * FROW + kk];
        }
        sacc[nf] = __builtin_amdgcn_wmma_f32_16x16x32_f16(
            false, qa[kf].h, false, bfr.h, (short)0, sacc[nf], false, false);
      }

    // scale + key mask (exactly reference: masked -> -1e6, then softmax)
    #pragma unroll
    for (int nf = 0; nf < 4; ++nf) {
      const int col = kv0 + nf * 16 + lrow;
      #pragma unroll
      for (int r = 0; r < 8; ++r)
        sacc[nf][r] = (col < vlen) ? sacc[nf][r] * 0.125f : MASKV;
    }

    // online softmax (rows split across 16-lane halves)
    float mnew[8];
    #pragma unroll
    for (int r = 0; r < 8; ++r) {
      float mx = fmaxf(fmaxf(sacc[0][r], sacc[1][r]), fmaxf(sacc[2][r], sacc[3][r]));
      #pragma unroll
      for (int d = 1; d < 16; d <<= 1) mx = fmaxf(mx, __shfl_xor(mx, d, 32));
      mnew[r] = fmaxf(mst[r], mx);
    }
    #pragma unroll
    for (int r = 0; r < 8; ++r) {
      const float fac = __expf(mst[r] - mnew[r]);
      lst[r] *= fac;
      #pragma unroll
      for (int df = 0; df < 4; ++df) oacc[df][r] *= fac;
      mst[r] = mnew[r];
    }
    #pragma unroll
    for (int nf = 0; nf < 4; ++nf)
      #pragma unroll
      for (int r = 0; r < 8; ++r)
        sacc[nf][r] = __expf(sacc[nf][r] - mnew[r]);
    #pragma unroll
    for (int r = 0; r < 8; ++r) {
      float rs = sacc[0][r] + sacc[1][r] + sacc[2][r] + sacc[3][r];
      #pragma unroll
      for (int d = 1; d < 16; d <<= 1) rs += __shfl_xor(rs, d, 32);
      lst[r] += rs;
    }

    // P: C-layout -> A-layout via per-wave LDS scratch (same-wave DS ops in order)
    #pragma unroll
    for (int nf = 0; nf < 4; ++nf)
      #pragma unroll
      for (int r = 0; r < 8; ++r)
        Ps[wave][(lhalf * 8 + r) * FROW + nf * 16 + lrow] = (_Float16)sacc[nf][r];
    asm volatile("s_wait_dscnt 0x0" ::: "memory");

    Frag16 pa[2];
    #pragma unroll
    for (int kf = 0; kf < 2; ++kf)
      #pragma unroll
      for (int v = 0; v < 8; ++v) {
        const int kk = kf * 32 + (v < 4 ? 2 * v : 16 + 2 * (v - 4)) + lhalf * 8;
        pa[kf].u[v] = *(const u32*)&Ps[wave][lrow * FROW + kk];
      }

    // O += P * V
    #pragma unroll
    for (int df = 0; df < 4; ++df)
      #pragma unroll
      for (int kf = 0; kf < 2; ++kf) {
        Frag16 bfr;
        #pragma unroll
        for (int v = 0; v < 8; ++v) {
          const int jj = kf * 32 + (v < 4 ? 2 * v : 16 + 2 * (v - 4)) + lhalf * 8;   // B[j][d] = VsT[d][j]
          bfr.u[v] = *(const u32*)&VsT[(df * 16 + lrow) * FROW + jj];
        }
        oacc[df] = __builtin_amdgcn_wmma_f32_16x16x32_f16(
            false, pa[kf].h, false, bfr.h, (short)0, oacc[df], false, false);
      }
  }

  // normalize + write head slice (concat-heads layout [B,S,D])
  #pragma unroll
  for (int r = 0; r < 8; ++r) {
    const float inv = 1.0f / lst[r];
    const long row = (long)(b * S_LEN + q0 + wave * 16 + lhalf * 8 + r);
    #pragma unroll
    for (int df = 0; df < 4; ++df)
      Op[row * DMODEL + headoff + df * 16 + lrow] = (_Float16)(oacc[df][r] * inv);
  }
}

// ---------------------------------------------------------------------------
extern "C" void kernel_launch(void* const* d_in, const int* in_sizes, int n_in,
                              void* d_out, int out_size, void* d_ws, size_t ws_size,
                              hipStream_t stream) {
  const float* queries = (const float*)d_in[0];
  const float* keys    = (const float*)d_in[1];
  const float* values  = (const float*)d_in[2];
  const int*   vlens   = (const int*)d_in[3];
  const float* Wq      = (const float*)d_in[4];
  const float* Wk      = (const float*)d_in[5];
  const float* Wv      = (const float*)d_in[6];
  const float* Wo      = (const float*)d_in[7];
  float* out = (float*)d_out;

  const size_t elems = (size_t)MROWS * DMODEL;   // 8M f16 = 16 MB each
  _Float16* Qh = (_Float16*)d_ws;
  _Float16* Kh = Qh + elems;
  _Float16* Vh = Kh + elems;
  _Float16* Oh = Vh + elems;                     // 64 MB total workspace

  dim3 gblk(256), ggrd(DMODEL / 128, MROWS / 128);      // 8 x 64
  gemm_wmma_kernel<float, _Float16><<<ggrd, gblk, 0, stream>>>(queries, Wq, Qh, MROWS, DMODEL, DMODEL);
  gemm_wmma_kernel<float, _Float16><<<ggrd, gblk, 0, stream>>>(keys,    Wk, Kh, MROWS, DMODEL, DMODEL);
  gemm_wmma_kernel<float, _Float16><<<ggrd, gblk, 0, stream>>>(values,  Wv, Vh, MROWS, DMODEL, DMODEL);

  dim3 fgrd(S_LEN / 64, BSZ * NHEAD);                   // 32 x 64 blocks
  flash_attn_kernel<<<fgrd, dim3(128), 0, stream>>>(Qh, Kh, Vh, vlens, Oh);

  gemm_wmma_kernel<_Float16, float><<<ggrd, gblk, 0, stream>>>(Oh, Wo, out, MROWS, DMODEL, DMODEL);
}
